// GraphEmbeddingGPS_4226247819267
// MI455X (gfx1250) — matrix operand order repaired and checked by reading, not compiled, not run on hardware
//
#include <hip/hip_runtime.h>
#include <hip/hip_bf16.h>

// ---------------- problem constants ----------------
#define TN_  32768      // B*N nodes
#define C_   256        // channels
#define Bg_  64         // graphs
#define Nn_  512        // nodes per graph
#define H_   4          // heads
#define DH_  64         // head dim
#define M_   266        // Performer features
#define L_   5          // layers
#define EPS_K   1e-3f
#define LN_EPS  1e-5f

// ---------------- WMMA types ----------------
typedef __bf16 bf16_t;
typedef __attribute__((ext_vector_type(16))) __bf16 v16bf;
typedef __attribute__((ext_vector_type(8)))  __bf16 v8bf;
typedef __attribute__((ext_vector_type(8)))  float  v8f;

#define BM 128
#define BN 64
#define BK 32
#define BKP 40   // padded bf16 LDS row pitch (80B: rows 16B aligned, conflict-free b128 frags)

__device__ __forceinline__ float gelu_exact(float x) {
    return 0.5f * x * (1.0f + erff(x * 0.70710678118654752440f));
}

// Generic batched strided GEMM: C[z] = act(A[z] @ B[z] + bias) * rscale_row
// B source: either fp32 strided (Bm, bZ, bR, bC) or pre-converted bf16 with
// contiguous-k rows (Bbf, row pitch btR, batch-invariant) — the fast path.
struct GemmP {
    const float*  A;
    const float*  Bm;
    const bf16_t* Bbf;
    float*        C;
    const float*  bias;    // per-col bias or nullptr
    const float*  rscale;  // per-row scale or nullptr
    long long aZ, aR, aC;
    long long bZ, bR, bC, btR;
    long long cZ, cR, cC;
    long long rZ;
    int M, N, K, act;      // act: 0 none, 1 gelu, 2 relu+EPS_K
};

__global__ __launch_bounds__(256) void gemm_wmma_k(GemmP p) {
    __shared__ __align__(16) bf16_t As[BM * BKP];
    __shared__ __align__(16) bf16_t Bs[BN * BKP];

    const int z = blockIdx.z;
    const float* A  = p.A + (long long)z * p.aZ;
    const float* Bm = p.Bm ? (p.Bm + (long long)z * p.bZ) : nullptr;
    float*       C  = p.C + (long long)z * p.cZ;

    const int m0 = blockIdx.x * BM;
    const int n0 = blockIdx.y * BN;
    const int tid  = threadIdx.x;
    const int wid  = tid >> 5;
    const int lane = tid & 31;
    const int wm = (wid & 3) * 32;   // wave row offset within tile
    const int wn = (wid >> 2) * 32;  // wave col offset within tile
    const int half = lane >> 4;      // 0: lanes 0-15, 1: lanes 16-31
    const int r    = lane & 15;

    v8f acc[2][2] = {};

    const int ktiles = (p.K + BK - 1) / BK;
    for (int kt = 0; kt < ktiles; ++kt) {
        const int k0 = kt * BK;

        // ---- stage A tile (BM x BK) ----
        if (p.aC == 1 && (k0 + BK) <= p.K) {
            // fast path: 8 k-contiguous floats per item, vector loads + packed cvt
            #pragma unroll
            for (int t = tid; t < BM * 4; t += 256) {
                int i = t >> 2, j = (t & 3) << 3;
                int gm = m0 + i;
                v8bf dv = {};
                if (gm < p.M) {
                    const float* s = A + (long long)gm * p.aR + (k0 + j);
                    float2 f0 = *(const float2*)(s);
                    float2 f1 = *(const float2*)(s + 2);
                    float2 f2 = *(const float2*)(s + 4);
                    float2 f3 = *(const float2*)(s + 6);
                    dv[0] = (bf16_t)f0.x; dv[1] = (bf16_t)f0.y;
                    dv[2] = (bf16_t)f1.x; dv[3] = (bf16_t)f1.y;
                    dv[4] = (bf16_t)f2.x; dv[5] = (bf16_t)f2.y;
                    dv[6] = (bf16_t)f3.x; dv[7] = (bf16_t)f3.y;
                    if (k0 + BK < p.K) __builtin_prefetch(s + BK, 0, 1);
                }
                *(v8bf*)&As[i * BKP + j] = dv;
            }
        } else {
            for (int t = tid; t < BM * BK; t += 256) {
                int i = t / BK, j = t % BK;
                int gm = m0 + i, gk = k0 + j;
                float v = (gm < p.M && gk < p.K)
                            ? A[(long long)gm * p.aR + (long long)gk * p.aC] : 0.0f;
                As[i * BKP + j] = (bf16_t)v;
            }
        }

        // ---- stage B tile (BN x BK), LDS layout [n][k] ----
        if (p.Bbf) {
            // fast path: bf16 rows already k-contiguous (weights^T / proj)
            #pragma unroll
            for (int t = tid; t < BN * 2; t += 256) {
                int i = t >> 1, j = (t & 1) << 4;
                int gn = n0 + i;
                v8bf c0 = {}, c1 = {};
                if (gn < p.N) {
                    const bf16_t* s = p.Bbf + (long long)gn * p.btR + (k0 + j);
                    c0 = *(const v8bf*)(s);
                    c1 = *(const v8bf*)(s + 8);
                }
                *(v8bf*)&Bs[i * BKP + j]     = c0;
                *(v8bf*)&Bs[i * BKP + j + 8] = c1;
            }
        } else {
            for (int t = tid; t < BN * BK; t += 256) {
                int i = t / BK, j = t % BK;
                int gn = n0 + i, gk = k0 + j;
                float v = (gn < p.N && gk < p.K)
                            ? Bm[(long long)gk * p.bR + (long long)gn * p.bC] : 0.0f;
                Bs[i * BKP + j] = (bf16_t)v;
            }
        }
        __syncthreads();

        #pragma unroll
        for (int i = 0; i < 2; ++i) {
            // A fragment, 16x32 bf16: lanes 0-15 row=r K{0..7,16..23}; lanes 16-31 K{8..15,24..31}
            const bf16_t* ap = &As[(wm + i * 16 + r) * BKP];
            v8bf a0 = *(const v8bf*)(ap + half * 8);
            v8bf a1 = *(const v8bf*)(ap + 16 + half * 8);
            v16bf av = __builtin_shufflevector(a0, a1, 0,1,2,3,4,5,6,7,8,9,10,11,12,13,14,15);
            #pragma unroll
            for (int j = 0; j < 2; ++j) {
                // B fragment, 32x16 bf16: lanes 0-15 col=r K{0..7,16..23}; lanes 16-31 K{8..15,24..31}
                const bf16_t* bp = &Bs[(wn + j * 16 + r) * BKP];
                v8bf b0 = *(const v8bf*)(bp + half * 8);
                v8bf b1 = *(const v8bf*)(bp + 16 + half * 8);
                v16bf bv = __builtin_shufflevector(b0, b1, 0,1,2,3,4,5,6,7,8,9,10,11,12,13,14,15);
                acc[i][j] = __builtin_amdgcn_wmma_f32_16x16x32_bf16(
                    false, av, false, bv, (short)0, acc[i][j], false, false);
            }
        }
        __syncthreads();
    }

    // epilogue: C/D layout — lane r covers col, VGPR v covers row v (+8 for upper lanes)
    #pragma unroll
    for (int i = 0; i < 2; ++i) {
        #pragma unroll
        for (int j = 0; j < 2; ++j) {
            int col = n0 + wn + j * 16 + r;
            if (col >= p.N) continue;
            float bv = p.bias ? p.bias[col] : 0.0f;
            #pragma unroll
            for (int v = 0; v < 8; ++v) {
                int row = m0 + wm + i * 16 + v + half * 8;
                if (row >= p.M) continue;
                float x = acc[i][j][v] + bv;
                if (p.act == 1)      x = gelu_exact(x);
                else if (p.act == 2) x = fmaxf(x, 0.0f) + EPS_K;
                if (p.rscale)        x *= p.rscale[(long long)z * p.rZ + row];
                C[(long long)row * p.cR + (long long)col * p.cC] = x;
            }
        }
    }
}

// ---------------- weight conversion kernels ----------------

// Wt[n*K + k] = (bf16) W[k*N + n]   (transpose + convert, run once)
__global__ __launch_bounds__(256) void wconv_t_k(const float* __restrict__ W,
                                                 bf16_t* __restrict__ Wt, int K, int N) {
    int idx = blockIdx.x * 256 + threadIdx.x;
    if (idx >= K * N) return;
    int n = idx / K, k = idx % K;
    Wt[idx] = (bf16_t)W[(long long)k * N + n];
}

// straight fp32 -> bf16 convert (proj is already k-contiguous per row)
__global__ __launch_bounds__(256) void cvt_bf16_k(const float* __restrict__ s,
                                                  bf16_t* __restrict__ d, long long n) {
    long long i = (long long)blockIdx.x * 256 + threadIdx.x;
    long long stride = (long long)gridDim.x * 256;
    for (; i < n; i += stride) d[i] = (bf16_t)s[i];
}

// ---------------- elementwise / graph kernels ----------------

__global__ __launch_bounds__(256) void node_embed_k(const float* __restrict__ x,
                                                    const float* __restrict__ w,
                                                    const float* __restrict__ b,
                                                    float* __restrict__ atoms) {
    int i = blockIdx.x, c = threadIdx.x;
    __shared__ float lx[11];
    if (c < 11) lx[c] = logf(x[(long long)i * 11 + c] + 1.0f);
    __syncthreads();
    float s = b[c];
    #pragma unroll
    for (int j = 0; j < 11; ++j) s += lx[j] * w[j * C_ + c];
    atoms[(long long)i * C_ + c] = s;
}

// msg = relu(atoms[src] + (edge_attr @ edge_w + edge_b)); scatter-add to aggr[dst]
__global__ __launch_bounds__(256) void edge_msg_k(const float* __restrict__ atoms,
                                                  const float* __restrict__ eattr,
                                                  const int* __restrict__ ei,
                                                  const float* __restrict__ ew,
                                                  const float* __restrict__ eb,
                                                  float* __restrict__ aggr, int E) {
    int e = blockIdx.x, c = threadIdx.x;
    int src = ei[e], dst = ei[E + e];
    long long e4 = (long long)e * 4;
    float a0 = eattr[e4], a1 = eattr[e4 + 1], a2 = eattr[e4 + 2], a3 = eattr[e4 + 3];
    float ea = eb[c] + a0 * ew[c] + a1 * ew[C_ + c] + a2 * ew[2 * C_ + c] + a3 * ew[3 * C_ + c];
    float m = atoms[(long long)src * C_ + c] + ea;
    if (m > 0.0f) atomicAdd(&aggr[(long long)dst * C_ + c], m);
}

__global__ __launch_bounds__(256) void add_k(const float* __restrict__ a,
                                             const float* __restrict__ b,
                                             float* __restrict__ o, long long n) {
    long long i = (long long)blockIdx.x * blockDim.x + threadIdx.x;
    long long stride = (long long)gridDim.x * blockDim.x;
    for (; i < n; i += stride) o[i] = a[i] + b[i];
}

// out = LN(a + b) over C=256 channels
__global__ __launch_bounds__(256) void ln_add_k(const float* __restrict__ a,
                                                const float* __restrict__ b,
                                                const float* __restrict__ g,
                                                const float* __restrict__ beta,
                                                float* __restrict__ out) {
    int row = blockIdx.x, c = threadIdx.x;
    long long off = (long long)row * C_ + c;
    float x = a[off] + b[off];
    __shared__ float red[256];
    red[c] = x; __syncthreads();
    for (int s = 128; s > 0; s >>= 1) { if (c < s) red[c] += red[c + s]; __syncthreads(); }
    float mu = red[0] * (1.0f / C_);
    __syncthreads();
    float d = x - mu;
    red[c] = d * d; __syncthreads();
    for (int s = 128; s > 0; s >>= 1) { if (c < s) red[c] += red[c + s]; __syncthreads(); }
    float var = red[0] * (1.0f / C_);
    out[off] = d * rsqrtf(var + LN_EPS) * g[c] + beta[c];
}

// ksum[b][m] = sum_n kpT[b][m][n]  — one wave per m, coalesced contiguous reads
__global__ __launch_bounds__(256) void ksum_k(const float* __restrict__ kpT,
                                              float* __restrict__ ks) {
    int b = blockIdx.x;
    int wid = threadIdx.x >> 5, lane = threadIdx.x & 31;
    int m = blockIdx.y * 8 + wid;
    if (m >= M_) return;
    const float* p = kpT + ((long long)b * M_ + m) * Nn_;
    float s = 0.0f;
    for (int n = lane; n < Nn_; n += 32) s += p[n];
    for (int off = 16; off > 0; off >>= 1) s += __shfl_down(s, off, 32);
    if (lane == 0) ks[b * M_ + m] = s;
}

// dinv[b][n] = 1 / dot(qp[b][n][:], ksum[b][:])  — one wave per n
__global__ __launch_bounds__(256) void dinv_k(const float* __restrict__ qp,
                                              const float* __restrict__ ks,
                                              float* __restrict__ dinv) {
    int b = blockIdx.x;
    int wid = threadIdx.x >> 5, lane = threadIdx.x & 31;
    int n = blockIdx.y * 8 + wid;
    const float* q = qp + ((long long)b * Nn_ + n) * M_;
    const float* s = ks + b * M_;
    float acc = 0.0f;
    for (int m = lane; m < M_; m += 32) acc += q[m] * s[m];
    for (int off = 16; off > 0; off >>= 1) acc += __shfl_down(acc, off, 32);
    if (lane == 0) dinv[(long long)b * Nn_ + n] = 1.0f / acc;
}

// global mean pool over N nodes per graph
__global__ __launch_bounds__(256) void pool_k(const float* __restrict__ atoms,
                                              float* __restrict__ out) {
    int b = blockIdx.x, c = threadIdx.x;
    float s = 0.0f;
    for (int n = 0; n < Nn_; ++n) s += atoms[((long long)b * Nn_ + n) * C_ + c];
    out[b * C_ + c] = s * (1.0f / Nn_);
}

// ---------------- host orchestration ----------------
static inline void run_gemm(hipStream_t st, const float* A, const float* B,
                            const bf16_t* Bbf, float* C,
                            const float* bias, const float* rscale,
                            long long aZ, long long aR, long long aC,
                            long long bZ, long long bR, long long bC, long long btR,
                            long long cZ, long long cR, long long cC,
                            long long rZ, int M, int N, int K, int act, int nz) {
    GemmP p{A, B, Bbf, C, bias, rscale, aZ, aR, aC, bZ, bR, bC, btR, cZ, cR, cC, rZ, M, N, K, act};
    dim3 grid((M + BM - 1) / BM, (N + BN - 1) / BN, nz);
    gemm_wmma_k<<<grid, 256, 0, st>>>(p);
}

extern "C" void kernel_launch(void* const* d_in, const int* in_sizes, int n_in,
                              void* d_out, int out_size, void* d_ws, size_t ws_size,
                              hipStream_t stream) {
    const float* x         = (const float*)d_in[0];
    const float* edge_attr = (const float*)d_in[1];
    const int*   edge_idx  = (const int*)  d_in[2];
    // d_in[3] = batch (unused; equal-sized graphs)
    const float* node_w = (const float*)d_in[4];
    const float* node_b = (const float*)d_in[5];
    const float* edge_w = (const float*)d_in[6];
    const float* edge_b = (const float*)d_in[7];
    const float* gine_w1 = (const float*)d_in[8];
    const float* gine_b1 = (const float*)d_in[9];
    const float* gine_w2 = (const float*)d_in[10];
    const float* gine_b2 = (const float*)d_in[11];
    const float* q_w = (const float*)d_in[12];
    const float* k_w = (const float*)d_in[13];
    const float* v_w = (const float*)d_in[14];
    const float* o_w = (const float*)d_in[15];
    const float* o_b = (const float*)d_in[16];
    const float* proj = (const float*)d_in[17];
    const float* n1g = (const float*)d_in[18];
    const float* n1b = (const float*)d_in[19];
    const float* n2g = (const float*)d_in[20];
    const float* n2b = (const float*)d_in[21];
    const float* n3g = (const float*)d_in[22];
    const float* n3b = (const float*)d_in[23];
    const float* mlp_w1 = (const float*)d_in[24];
    const float* mlp_b1 = (const float*)d_in[25];
    const float* mlp_w2 = (const float*)d_in[26];
    const float* mlp_b2 = (const float*)d_in[27];
    float* out = (float*)d_out;

    const int E = in_sizes[2] / 2;
    const long long TNC = (long long)TN_ * C_;
    const long long QPSZ = (long long)Bg_ * Nn_ * M_;
    const long long WSQ  = (long long)C_ * C_;      // 65536

    // ---- workspace carving: fp32 region ----
    float* ws    = (float*)d_ws;
    float* atoms = ws;  ws += TNC;
    float* aggr  = ws;  ws += TNC;
    float* bufA  = ws;  ws += TNC;   // gine input sum / h_att (LN2 out)
    float* bufB  = ws;  ws += TNC;   // gine t1 / att output
    float* bufC  = ws;  ws += TNC;   // gine h / o-proj out / mlp out
    float* hloc  = ws;  ws += TNC;
    float* bufO  = ws;  ws += TNC;
    float* bq    = ws;  ws += TNC;
    float* bk    = ws;  ws += TNC;
    float* bv    = ws;  ws += TNC;
    float* t2    = ws;  ws += 2 * TNC;
    float* qp    = ws;  ws += QPSZ;                 // [b][n][m]
    float* kpT   = ws;  ws += QPSZ;                 // [b][m][n]  (stored transposed)
    float* ctx   = ws;  ws += (long long)Bg_ * M_ * DH_;
    float* ksum  = ws;  ws += (long long)Bg_ * M_;
    float* dinv  = ws;  ws += (long long)Bg_ * Nn_;

    // ---- bf16 region: transposed weights [n][k] + proj ----
    bf16_t* wsb = (bf16_t*)ws;
    bf16_t* wt_g1 = wsb;  wsb += L_ * WSQ;
    bf16_t* wt_g2 = wsb;  wsb += L_ * WSQ;
    bf16_t* wt_q  = wsb;  wsb += L_ * WSQ;
    bf16_t* wt_k  = wsb;  wsb += L_ * WSQ;
    bf16_t* wt_v  = wsb;  wsb += L_ * WSQ;
    bf16_t* wt_o  = wsb;  wsb += L_ * WSQ;
    bf16_t* wt_m1 = wsb;  wsb += L_ * 2 * WSQ;      // [512][256]
    bf16_t* wt_m2 = wsb;  wsb += L_ * 2 * WSQ;      // [256][512]
    bf16_t* prj_b = wsb;  wsb += (long long)L_ * M_ * DH_;  // [m][d], k-contiguous

    // ---- one-time weight conversion ----
    const int GW = (C_ * C_ + 255) / 256;           // 256 blocks
    for (int l = 0; l < L_; ++l) {
        long long o1 = (long long)l * WSQ, o2 = (long long)l * 2 * WSQ;
        wconv_t_k<<<GW, 256, 0, stream>>>(gine_w1 + o1, wt_g1 + o1, C_, C_);
        wconv_t_k<<<GW, 256, 0, stream>>>(gine_w2 + o1, wt_g2 + o1, C_, C_);
        wconv_t_k<<<GW, 256, 0, stream>>>(q_w + o1, wt_q + o1, C_, C_);
        wconv_t_k<<<GW, 256, 0, stream>>>(k_w + o1, wt_k + o1, C_, C_);
        wconv_t_k<<<GW, 256, 0, stream>>>(v_w + o1, wt_v + o1, C_, C_);
        wconv_t_k<<<GW, 256, 0, stream>>>(o_w + o1, wt_o + o1, C_, C_);
        wconv_t_k<<<2 * GW, 256, 0, stream>>>(mlp_w1 + o2, wt_m1 + o2, C_, 2 * C_);
        wconv_t_k<<<2 * GW, 256, 0, stream>>>(mlp_w2 + o2, wt_m2 + o2, 2 * C_, C_);
    }
    cvt_bf16_k<<<512, 256, 0, stream>>>(proj, prj_b, (long long)L_ * M_ * DH_);

    // atoms = log(x+1) @ node_w + node_b
    node_embed_k<<<TN_, 256, 0, stream>>>(x, node_w, node_b, atoms);

    for (int l = 0; l < L_; ++l) {
        // ---- GINEConv ----
        hipMemsetAsync(aggr, 0, TNC * sizeof(float), stream);
        edge_msg_k<<<E, 256, 0, stream>>>(atoms, edge_attr, edge_idx, edge_w, edge_b, aggr, E);
        add_k<<<2048, 256, 0, stream>>>(atoms, aggr, bufA, TNC);
        run_gemm(stream, bufA, nullptr, wt_g1 + (long long)l * WSQ, bufB,
                 gine_b1 + l * C_, nullptr, 0, C_, 1, 0, 0, 0, C_, 0, C_, 1, 0,
                 TN_, C_, C_, /*gelu*/1, 1);
        run_gemm(stream, bufB, nullptr, wt_g2 + (long long)l * WSQ, bufC,
                 gine_b2 + l * C_, nullptr, 0, C_, 1, 0, 0, 0, C_, 0, C_, 1, 0,
                 TN_, C_, C_, 0, 1);
        ln_add_k<<<TN_, 256, 0, stream>>>(bufC, atoms, n1g + l * C_, n1b + l * C_, hloc);

        // ---- Performer attention ----
        run_gemm(stream, atoms, nullptr, wt_q + (long long)l * WSQ, bq, nullptr, nullptr,
                 0, C_, 1, 0, 0, 0, C_, 0, C_, 1, 0, TN_, C_, C_, 0, 1);
        run_gemm(stream, atoms, nullptr, wt_k + (long long)l * WSQ, bk, nullptr, nullptr,
                 0, C_, 1, 0, 0, 0, C_, 0, C_, 1, 0, TN_, C_, C_, 0, 1);
        run_gemm(stream, atoms, nullptr, wt_v + (long long)l * WSQ, bv, nullptr, nullptr,
                 0, C_, 1, 0, 0, 0, C_, 0, C_, 1, 0, TN_, C_, C_, 0, 1);

        const bf16_t* prj = prj_b + (long long)l * M_ * DH_;
        for (int h = 0; h < H_; ++h) {
            // qp[b][n][m] = relu(q . proj[m]) + eps
            run_gemm(stream, bq + h * DH_, nullptr, prj, qp, nullptr, nullptr,
                     (long long)Nn_ * C_, C_, 1,
                     0, 0, 0, DH_,
                     (long long)Nn_ * M_, M_, 1, 0,
                     Nn_, M_, DH_, /*relu+eps*/2, Bg_);
            // kpT[b][m][n] via transposed store (cR=1, cC=Nn)
            run_gemm(stream, bk + h * DH_, nullptr, prj, kpT, nullptr, nullptr,
                     (long long)Nn_ * C_, C_, 1,
                     0, 0, 0, DH_,
                     (long long)M_ * Nn_, 1, Nn_, 0,
                     Nn_, M_, DH_, 2, Bg_);
            ksum_k<<<dim3(Bg_, (M_ + 7) / 8), 256, 0, stream>>>(kpT, ksum);
            dinv_k<<<dim3(Bg_, Nn_ / 8), 256, 0, stream>>>(qp, ksum, dinv);
            // ctx[b][m][d] = sum_n kpT[b][m][n] * v[b][n][h][d]  (A contiguous)
            run_gemm(stream, kpT, bv + h * DH_, nullptr, ctx, nullptr, nullptr,
                     (long long)M_ * Nn_, Nn_, 1,
                     (long long)Nn_ * C_, C_, 1, 0,     // B batched per graph: bZ = Nn*C
                     (long long)M_ * DH_, DH_, 1, 0,
                     M_, DH_, Nn_, 0, Bg_);
            // att[b][n][h][d] = dinv[b][n] * sum_m qp[b][n][m] * ctx[b][m][d]
            run_gemm(stream, qp, ctx, nullptr, bufB + h * DH_, nullptr, dinv,
                     (long long)Nn_ * M_, M_, 1,
                     (long long)M_ * DH_, DH_, 1, 0,    // B batched per graph: bZ = M*DH
                     (long long)Nn_ * C_, C_, 1,
                     Nn_, Nn_, DH_, M_, 0, Bg_);
        }
        // h_att = att @ o_w + o_b ; LN2(h_att + atoms)
        run_gemm(stream, bufB, nullptr, wt_o + (long long)l * WSQ, bufC,
                 o_b + l * C_, nullptr, 0, C_, 1, 0, 0, 0, C_, 0, C_, 1, 0,
                 TN_, C_, C_, 0, 1);
        ln_add_k<<<TN_, 256, 0, stream>>>(bufC, atoms, n2g + l * C_, n2b + l * C_, bufA);

        // ---- combine + MLP + final norm ----
        add_k<<<2048, 256, 0, stream>>>(hloc, bufA, bufO, TNC);
        run_gemm(stream, bufO, nullptr, wt_m1 + (long long)l * 2 * WSQ, t2,
                 mlp_b1 + l * 2 * C_, nullptr, 0, C_, 1, 0, 0, 0, C_, 0, 2 * C_, 1, 0,
                 TN_, 2 * C_, C_, /*gelu*/1, 1);
        run_gemm(stream, t2, nullptr, wt_m2 + (long long)l * 2 * WSQ, bufC,
                 mlp_b2 + l * C_, nullptr, 0, 2 * C_, 1, 0, 0, 0, 2 * C_, 0, C_, 1, 0,
                 TN_, C_, 2 * C_, 0, 1);
        ln_add_k<<<TN_, 256, 0, stream>>>(bufO, bufC, n3g + l * C_, n3b + l * C_, atoms);
    }

    pool_k<<<Bg_, 256, 0, stream>>>(atoms, out);
    (void)n_in; (void)out_size; (void)ws_size;
}
